// CausalSelfAttention_39582418599986
// MI455X (gfx1250) — matrix operand order repaired and checked by reading, not compiled, not run on hardware
//
#include <hip/hip_runtime.h>
#include <hip/hip_bf16.h>
#include <stdint.h>

#define B_ 2
#define T_ 2048
#define C_ 1024
#define H_ 16
#define HD_ 64
#define F3_ (3 * C_)

typedef __attribute__((ext_vector_type(16))) _Float16 v16h;
typedef __attribute__((ext_vector_type(8)))  _Float16 v8h;
typedef __attribute__((ext_vector_type(8)))  float    v8f;
typedef __attribute__((vector_size(16)))     int      v4i;

#if defined(__has_builtin)
#  if __has_builtin(__builtin_amdgcn_global_load_async_to_lds_b128) && \
      __has_builtin(__builtin_amdgcn_s_wait_asynccnt)
#    define USE_ASYNC 1
#  endif
#endif
#ifndef USE_ASYNC
#  define USE_ASYNC 0
#endif

__device__ __forceinline__ v8f wmma_f16(v16h a, v16h b, v8f c) {
    return __builtin_amdgcn_wmma_f32_16x16x32_f16(false, a, false, b, (short)0, c, false, false);
}

#if USE_ASYNC
// 16B per lane, global -> LDS, tracked by ASYNCcnt
__device__ __forceinline__ void async_copy16(const _Float16* g, _Float16* l) {
    __builtin_amdgcn_global_load_async_to_lds_b128(
        (__attribute__((address_space(1))) v4i*)(uintptr_t)g,
        (__attribute__((address_space(3))) v4i*)(unsigned)(uintptr_t)l,
        0, 0);
}
#endif

// A-fragment (16x32, MxK) from row-major [16 rows, ld] f16.
__device__ __forceinline__ v16h load_fragA(const _Float16* p, int ld) {
    int lane = threadIdx.x & 31;
    const _Float16* q = p + (size_t)(lane & 15) * ld + ((lane >> 4) << 3);
    v8h lo = *reinterpret_cast<const v8h*>(q);
    v8h hi = *reinterpret_cast<const v8h*>(q + 16);
    return __builtin_shufflevector(lo, hi, 0,1,2,3,4,5,6,7,8,9,10,11,12,13,14,15);
}

// B-fragment (32x16, KxN) from row-major [N=16 rows, ld] f16 (NT layout).
__device__ __forceinline__ v16h load_fragB(const _Float16* p, int ld) {
    int lane = threadIdx.x & 31;
    const _Float16* q = p + (size_t)(lane & 15) * ld + ((lane >> 4) << 3);
    v8h lo = *reinterpret_cast<const v8h*>(q);
    v8h hi = *reinterpret_cast<const v8h*>(q + 16);
    return __builtin_shufflevector(lo, hi, 0,1,2,3,4,5,6,7,8,9,10,11,12,13,14,15);
}

// ---------------- f32 -> f16 convert ----------------
__global__ __launch_bounds__(256) void cvt_f32_f16(const float* __restrict__ in,
                                                   _Float16* __restrict__ out, int n) {
    int i = blockIdx.x * 256 + threadIdx.x;
    if (i < n) out[i] = (_Float16)in[i];
}

// ---------------- NT GEMM: C[M,N] = A[M,K] * B[N,K]^T ----------------
// 8 waves/block; block tile 256M x 64N; wave tile 32M x 64N (2 A frags share
// each B frag). B tile (64x32) staged in LDS via async double buffer; last
// K-step peeled so the steady-state loop is branch-free.
__global__ __launch_bounds__(256) void gemm_nt_f16(const _Float16* __restrict__ A,
                                                   const _Float16* __restrict__ Bm,
                                                   _Float16* __restrict__ Ch,
                                                   float* __restrict__ Cf,
                                                   int M, int N, int K) {
    __shared__ _Float16 bsh[2][64 * 32];
    int tid = threadIdx.x;
    int wave = tid >> 5;
    int m0 = (blockIdx.x * 8 + wave) * 32;
    int n0 = blockIdx.y * 64;

    // cooperative B staging: thread -> (row r of 64, 8-half chunk kc)
    int r = tid >> 2;
    int kc = (tid & 3) << 3;
    const _Float16* bsrc = Bm + (size_t)(n0 + r) * K + kc;
    _Float16* bdst0 = &bsh[0][r * 32 + kc];
    _Float16* bdst1 = &bsh[1][r * 32 + kc];

    int ntiles = K / 32;
    const _Float16* arow0 = A + (size_t)m0 * K;
    const _Float16* arow1 = arow0 + (size_t)16 * K;

#if USE_ASYNC
    async_copy16(bsrc, bdst0);
#else
    v8h breg = *reinterpret_cast<const v8h*>(bsrc);
#endif
    v16h a0 = load_fragA(arow0, K);
    v16h a1 = load_fragA(arow1, K);
    v8f acc0[4] = {}, acc1[4] = {};

    for (int i = 0; i < ntiles - 1; ++i) {
        _Float16* cur = (i & 1) ? &bsh[1][0] : &bsh[0][0];
#if USE_ASYNC
        async_copy16(bsrc + (i + 1) * 32, (i & 1) ? bdst0 : bdst1);
        __builtin_amdgcn_s_wait_asynccnt(1);
#else
        *reinterpret_cast<v8h*>((i & 1) ? bdst1 : bdst0) = breg;
        breg = *reinterpret_cast<const v8h*>(bsrc + (i + 1) * 32);
#endif
        __syncthreads();
        v16h a0n = load_fragA(arow0 + (i + 1) * 32, K);
        v16h a1n = load_fragA(arow1 + (i + 1) * 32, K);
#pragma unroll
        for (int t = 0; t < 4; ++t) {
            v16h b = load_fragB(cur + (t * 16) * 32, 32);
            acc0[t] = wmma_f16(a0, b, acc0[t]);
            acc1[t] = wmma_f16(a1, b, acc1[t]);
        }
        a0 = a0n; a1 = a1n;
        __syncthreads();
    }
    {   // peeled last K-step
        _Float16* cur = ((ntiles - 1) & 1) ? &bsh[1][0] : &bsh[0][0];
#if USE_ASYNC
        __builtin_amdgcn_s_wait_asynccnt(0);
#else
        *reinterpret_cast<v8h*>(((ntiles - 1) & 1) ? bdst1 : bdst0) = breg;
#endif
        __syncthreads();
#pragma unroll
        for (int t = 0; t < 4; ++t) {
            v16h b = load_fragB(cur + (t * 16) * 32, 32);
            acc0[t] = wmma_f16(a0, b, acc0[t]);
            acc1[t] = wmma_f16(a1, b, acc1[t]);
        }
    }

    int lane = tid & 31;
    int n = lane & 15, mb = (lane >> 4) << 3;
#pragma unroll
    for (int t = 0; t < 4; ++t) {
#pragma unroll
        for (int rr = 0; rr < 8; ++rr) {
            size_t i0 = (size_t)(m0 + mb + rr) * N + n0 + t * 16 + n;
            size_t i1 = (size_t)(m0 + 16 + mb + rr) * N + n0 + t * 16 + n;
            if (Ch) { Ch[i0] = (_Float16)acc0[t][rr]; Ch[i1] = (_Float16)acc1[t][rr]; }
            else    { Cf[i0] = acc0[t][rr];           Cf[i1] = acc1[t][rr]; }
        }
    }
}

// ---------------- RoPE on q,k (in place, f16) + write V^T [b,h,d,t] ----------------
__global__ __launch_bounds__(256) void rope_and_vt(_Float16* __restrict__ qkvh,
                                                   const float* __restrict__ fcos,
                                                   const float* __restrict__ fsin,
                                                   _Float16* __restrict__ vt) {
    int tid = blockIdx.x * 256 + threadIdx.x;   // B*T*H*32 threads
    int i = tid & 31;
    int h = (tid >> 5) & (H_ - 1);
    int t = (tid >> 9) & (T_ - 1);
    int b = tid >> 20;
    size_t row = (size_t)(b * T_ + t) * F3_;
    float c = fcos[t * 32 + i], s = fsin[t * 32 + i];

    _Float16* qp = qkvh + row + h * HD_ + 2 * i;
    float qe = (float)qp[0], qo = (float)qp[1];
    qp[0] = (_Float16)(qe * c - qo * s);
    qp[1] = (_Float16)(qe * s + qo * c);

    _Float16* kp = qkvh + row + C_ + h * HD_ + 2 * i;
    float ke = (float)kp[0], ko = (float)kp[1];
    kp[0] = (_Float16)(ke * c - ko * s);
    kp[1] = (_Float16)(ke * s + ko * c);

    const _Float16* vp = qkvh + row + 2 * C_ + h * HD_ + 2 * i;
    size_t vb = ((size_t)(b * H_ + h) * HD_ + 2 * i) * T_ + t;
    vt[vb] = vp[0];
    vt[vb + T_] = vp[1];
}

// ---------------- Flash attention: 1 wave per (q-tile 16, head, batch) ----------------
__device__ __forceinline__ void load_kfrags(v16h* bk, const _Float16* kptr, int j) {
    const _Float16* p0 = kptr + (size_t)j * F3_;
    bk[0] = load_fragB(p0, F3_);
    bk[1] = load_fragB(p0 + 32, F3_);
    bk[2] = load_fragB(p0 + (size_t)16 * F3_, F3_);
    bk[3] = load_fragB(p0 + (size_t)16 * F3_ + 32, F3_);
}

__global__ __launch_bounds__(32) void flash_attn(const _Float16* __restrict__ qkvh,
                                                 const _Float16* __restrict__ vt,
                                                 _Float16* __restrict__ yh) {
    __shared__ _Float16 pbuf[16 * 32];
    int qb = blockIdx.x * 16;
    int h = blockIdx.y, b = blockIdx.z;
    int lane = threadIdx.x & 31;
    int n = lane & 15, mb = (lane >> 4) << 3;

    const _Float16* qptr = qkvh + (size_t)(b * T_ + qb) * F3_ + h * HD_;
    v16h aq0 = load_fragA(qptr, F3_);
    v16h aq1 = load_fragA(qptr + 32, F3_);
    // fold softmax scale 1/sqrt(HD) into Q fragments
    aq0 = aq0 * (_Float16)0.125f;
    aq1 = aq1 * (_Float16)0.125f;
    const _Float16* kptr = qkvh + (size_t)b * T_ * F3_ + C_ + h * HD_;
    const _Float16* vtp = vt + (size_t)(b * H_ + h) * HD_ * T_;

    float mrow[8], lrow[8];
    v8f yacc[4] = {};
#pragma unroll
    for (int rr = 0; rr < 8; ++rr) { mrow[rr] = -3.0e38f; lrow[rr] = 0.0f; }

    v16h bk[4], bkn[4];
    load_kfrags(bk, kptr, 0);

    for (int j = 0; j <= qb; j += 32) {
        // V fragments for this chunk (independent of softmax -> overlap)
        v16h bv[4];
#pragma unroll
        for (int t = 0; t < 4; ++t) bv[t] = load_fragB(vtp + (size_t)(t * 16) * T_ + j, T_);
        // branchless prefetch of next chunk's K frags (clamped, always legal)
        int jn = (j + 32 <= qb) ? (j + 32) : j;
        load_kfrags(bkn, kptr, jn);

        v8f s0 = {}, s1 = {};
        s0 = wmma_f16(aq0, bk[0], s0);
        s0 = wmma_f16(aq1, bk[1], s0);
        s1 = wmma_f16(aq0, bk[2], s1);
        s1 = wmma_f16(aq1, bk[3], s1);

        // online softmax; rows of the C-fragment are lane-local across 4 y-tiles
#pragma unroll
        for (int rr = 0; rr < 8; ++rr) {
            int qi = qb + mb + rr;
            float e0 = (j + n <= qi)      ? s0[rr] : -3.0e38f;
            float e1 = (j + 16 + n <= qi) ? s1[rr] : -3.0e38f;
            float tmax = fmaxf(e0, e1);
            for (int msk = 8; msk >= 1; msk >>= 1)
                tmax = fmaxf(tmax, __shfl_xor(tmax, msk, 32));
            float nm = fmaxf(mrow[rr], tmax);
            float fac = __expf(mrow[rr] - nm);
            float p0 = __expf(e0 - nm);
            float p1 = __expf(e1 - nm);
            float ps = p0 + p1;
            for (int msk = 8; msk >= 1; msk >>= 1)
                ps += __shfl_xor(ps, msk, 32);
            lrow[rr] = lrow[rr] * fac + ps;
            mrow[rr] = nm;
#pragma unroll
            for (int t = 0; t < 4; ++t) yacc[t][rr] *= fac;
            pbuf[(mb + rr) * 32 + n] = (_Float16)p0;
            pbuf[(mb + rr) * 32 + 16 + n] = (_Float16)p1;
        }
        __syncthreads();
        v16h ap = load_fragA(pbuf, 32);
#pragma unroll
        for (int t = 0; t < 4; ++t) yacc[t] = wmma_f16(ap, bv[t], yacc[t]);
        __syncthreads();
#pragma unroll
        for (int t = 0; t < 4; ++t) bk[t] = bkn[t];
    }

    _Float16* yrow = yh + (size_t)(b * T_ + qb) * C_ + h * HD_;
#pragma unroll
    for (int t = 0; t < 4; ++t)
#pragma unroll
        for (int rr = 0; rr < 8; ++rr)
            yrow[(size_t)(mb + rr) * C_ + t * 16 + n] = (_Float16)(yacc[t][rr] / lrow[rr]);
}

extern "C" void kernel_launch(void* const* d_in, const int* in_sizes, int n_in,
                              void* d_out, int out_size, void* d_ws, size_t ws_size,
                              hipStream_t stream) {
    const float* x      = (const float*)d_in[0];
    const float* w_attn = (const float*)d_in[1];
    const float* w_proj = (const float*)d_in[2];
    const float* fcos   = (const float*)d_in[3];
    const float* fsin   = (const float*)d_in[4];
    float* out = (float*)d_out;

    char* ws = (char*)d_ws;
    _Float16* xh   = (_Float16*)(ws);                              //  8 MB [4096,1024]
    _Float16* wah  = (_Float16*)(ws + ((size_t)8  << 20));         //  6 MB [3072,1024]
    _Float16* wph  = (_Float16*)(ws + ((size_t)14 << 20));         //  2 MB [1024,1024]
    _Float16* qkvh = (_Float16*)(ws + ((size_t)16 << 20));         // 24 MB [4096,3072]
    _Float16* vt   = (_Float16*)(ws + ((size_t)40 << 20));         //  8 MB [B,H,64,T]
    _Float16* yh   = (_Float16*)(ws + ((size_t)48 << 20));         //  8 MB [4096,1024]

    cvt_f32_f16<<<(B_ * T_ * C_) / 256, 256, 0, stream>>>(x, xh, B_ * T_ * C_);
    cvt_f32_f16<<<(F3_ * C_) / 256, 256, 0, stream>>>(w_attn, wah, F3_ * C_);
    cvt_f32_f16<<<(C_ * C_) / 256, 256, 0, stream>>>(w_proj, wph, C_ * C_);

    dim3 g1((B_ * T_) / 256, F3_ / 64);
    gemm_nt_f16<<<g1, 256, 0, stream>>>(xh, wah, qkvh, nullptr, B_ * T_, F3_, C_);

    rope_and_vt<<<(B_ * T_ * H_ * 32) / 256, 256, 0, stream>>>(qkvh, fcos, fsin, vt);

    dim3 g3(T_ / 16, H_, B_);
    flash_attn<<<g3, 32, 0, stream>>>(qkvh, vt, yh);

    dim3 g4((B_ * T_) / 256, C_ / 64);
    gemm_nt_f16<<<g4, 256, 0, stream>>>(yh, wph, nullptr, out, B_ * T_, C_, C_);
}
